// MambaForecaster_83726092468228
// MI455X (gfx1250) — matrix operand order repaired and checked by reading, not compiled, not run on hardware
//
#include <hip/hip_runtime.h>
#include <hip/hip_bf16.h>
#include <hip/hip_fp16.h>

// ---------------- model dims ----------------
#define IN_DIM    32
#define D_MODEL   512
#define N_LAYERS  4
#define D_STATE   16
#define HORIZON   24
#define NUM_CITIES 32
#define D_CONV    4
#define D_INNER   1024          // EXPAND * D_MODEL
#define DT_RANK   32            // ceil(512/16)
#define BATCH     4
#define SEQLEN    512
#define MROWS     (BATCH * SEQLEN)   // 2048
#define HEAD_OUT  (HORIZON * NUM_CITIES) // 768

typedef __attribute__((ext_vector_type(16))) _Float16     v16h;
typedef __attribute__((ext_vector_type(8)))  float        v8f;
typedef __attribute__((ext_vector_type(4)))  unsigned int v4u;

// =====================================================================
// WMMA GEMM:  C[M,N] = A[Mpad,K] (f16, row-major) * W[N,K]^T (f16)
//             (+ bias[N]) (+ residual[M,N])
// One wave32 per 16x32 output tile: two 16x16 WMMA accumulators share
// one A fragment per K-step (K stepped by 32 per v_wmma pair).
// Caller guarantees A has ceil(M/16)*16 rows (zero-padded if needed),
// so the inner loop is branch-free with full EXEC; only stores guard M.
// =====================================================================
__global__ void __launch_bounds__(128)
wmma_gemm_f16(const _Float16* __restrict__ A, const _Float16* __restrict__ W,
              float* __restrict__ C, const float* __restrict__ bias,
              const float* __restrict__ residual, int M, int N, int K) {
  const int lane    = threadIdx.x & 31;
  const int wave    = blockIdx.x * (blockDim.x >> 5) + (threadIdx.x >> 5);
  const int n2tiles = N >> 5;                  // 32-wide column blocks
  const int mtiles  = (M + 15) >> 4;
  if (wave >= mtiles * n2tiles) return;        // wave-uniform exit
  const int tm = wave / n2tiles;
  const int tn = (wave - tm * n2tiles) * 2;    // first 16-col tile
  const int g  = lane >> 4;                    // lane half: 0 or 1
  const int li = lane & 15;

  const _Float16* __restrict__ arow  = A + (size_t)(tm * 16 + li) * K;
  const _Float16* __restrict__ wrow0 = W + (size_t)(tn * 16 + li) * K;
  const _Float16* __restrict__ wrow1 = wrow0 + (size_t)16 * K;

  v8f acc0 = {}, acc1 = {};
  for (int k0 = 0; k0 < K; k0 += 32) {
    union { v4u q[2]; v16h h; } af, bf0, bf1;
    // A 16x32 fragment: lane = row, halves {16g..16g+15} (K-contiguous)
    af.q[0]  = *(const v4u*)(arow  + k0 + 16 * g);
    af.q[1]  = *(const v4u*)(arow  + k0 + 16 * g + 8);
    // B 32x16 fragment: lane = column, halves {16g..16g+15} (K-contiguous)
    bf0.q[0] = *(const v4u*)(wrow0 + k0 + 16 * g);
    bf0.q[1] = *(const v4u*)(wrow0 + k0 + 16 * g + 8);
    bf1.q[0] = *(const v4u*)(wrow1 + k0 + 16 * g);
    bf1.q[1] = *(const v4u*)(wrow1 + k0 + 16 * g + 8);

    acc0 = __builtin_amdgcn_wmma_f32_16x16x32_f16(false, af.h, false, bf0.h,
                                                  (short)0, acc0, false, false);
    acc1 = __builtin_amdgcn_wmma_f32_16x16x32_f16(false, af.h, false, bf1.h,
                                                  (short)0, acc1, false, false);
  }

  const int   n0 = tn * 16 + li;
  const float b0 = bias ? bias[n0]      : 0.0f;
  const float b1 = bias ? bias[n0 + 16] : 0.0f;
#pragma unroll
  for (int r = 0; r < 8; ++r) {
    const int m = tm * 16 + r + 8 * g;         // C/D layout: VGPR r -> row r+8g
    if (m < M) {
      float v0 = acc0[r] + b0;
      float v1 = acc1[r] + b1;
      if (residual) {
        v0 += residual[(size_t)m * N + n0];
        v1 += residual[(size_t)m * N + n0 + 16];
      }
      C[(size_t)m * N + n0]      = v0;
      C[(size_t)m * N + n0 + 16] = v1;
    }
  }
}

// =====================================================================
// Elementwise / support kernels
// =====================================================================
__global__ void f32_to_f16_kernel(const float* __restrict__ in,
                                  _Float16* __restrict__ out, int n) {
  int i = blockIdx.x * blockDim.x + threadIdx.x;
  if (i < n) out[i] = (_Float16)in[i];
}

// x (B, IN_DIM, L) -> u16 (B*L, IN_DIM) f16
__global__ void prepare_u_kernel(const float* __restrict__ x,
                                 _Float16* __restrict__ u16) {
  int idx = blockIdx.x * blockDim.x + threadIdx.x;   // B*L*IN_DIM = 65536
  if (idx >= BATCH * SEQLEN * IN_DIM) return;
  int i = idx & (IN_DIM - 1);
  int l = (idx >> 5) & (SEQLEN - 1);
  int b = idx >> 14;                                  // /(512*32)
  u16[idx] = (_Float16)x[((size_t)b * IN_DIM + i) * SEQLEN + l];
}

// rmsnorm over D_MODEL, output f16 (GEMM A operand)
__global__ void __launch_bounds__(256)
rmsnorm_kernel(const float* __restrict__ h, const float* __restrict__ w,
               _Float16* __restrict__ out) {
  __shared__ float red[256];
  const int row = blockIdx.x;
  const float* hr = h + (size_t)row * D_MODEL;
  float s = 0.f;
  for (int c = threadIdx.x; c < D_MODEL; c += 256) { float v = hr[c]; s += v * v; }
  red[threadIdx.x] = s;
  __syncthreads();
  for (int off = 128; off > 0; off >>= 1) {
    if ((int)threadIdx.x < off) red[threadIdx.x] += red[threadIdx.x + off];
    __syncthreads();
  }
  const float scale = rsqrtf(red[0] / (float)D_MODEL + 1e-5f);
  for (int c = threadIdx.x; c < D_MODEL; c += 256)
    out[(size_t)row * D_MODEL + c] = (_Float16)(hr[c] * scale * w[c]);
}

// causal depthwise conv (D_CONV=4) + SiLU on xz[:, :D_INNER]
__global__ void conv_silu_kernel(const float* __restrict__ xz,
                                 const float* __restrict__ cw,
                                 const float* __restrict__ cb,
                                 float* __restrict__ xbs,
                                 _Float16* __restrict__ xb16) {
  int idx = blockIdx.x * blockDim.x + threadIdx.x;   // M * D_INNER = 2M
  if (idx >= MROWS * D_INNER) return;
  const int e = idx & (D_INNER - 1);
  const int m = idx >> 10;
  const int l = m & (SEQLEN - 1);
  const int b = m >> 9;
  float acc = cb[e];
#pragma unroll
  for (int j = 0; j < D_CONV; ++j) {
    const int ll = l - (D_CONV - 1) + j;
    if (ll >= 0)
      acc += xz[(size_t)(b * SEQLEN + ll) * (2 * D_INNER) + e] * cw[e * D_CONV + j];
  }
  const float s = acc / (1.0f + expf(-acc));  // silu
  xbs[idx]  = s;
  xb16[idx] = (_Float16)s;
}

// first DT_RANK columns of dbc (M,64) -> f16 (M,32) for dt_proj GEMM
__global__ void slice_dt_kernel(const float* __restrict__ dbc,
                                _Float16* __restrict__ out) {
  int idx = blockIdx.x * blockDim.x + threadIdx.x;   // M*32
  if (idx >= MROWS * DT_RANK) return;
  const int m = idx >> 5, r = idx & 31;
  out[idx] = (_Float16)dbc[(size_t)m * (DT_RANK + 2 * D_STATE) + r];
}

// selective scan: one thread per (b, e) channel; 16-wide state in registers.
__global__ void __launch_bounds__(256)
scan_kernel(const float* __restrict__ dtraw, const float* __restrict__ xbs,
            const float* __restrict__ dbc,   const float* __restrict__ xz,
            const float* __restrict__ A_log, const float* __restrict__ Dp,
            _Float16* __restrict__ y16) {
  int idx = blockIdx.x * blockDim.x + threadIdx.x;   // BATCH * D_INNER = 4096
  if (idx >= BATCH * D_INNER) return;
  const int b = idx >> 10;
  const int e = idx & (D_INNER - 1);
  float Ae[D_STATE], hs[D_STATE];
#pragma unroll
  for (int n = 0; n < D_STATE; ++n) {
    Ae[n] = -expf(A_log[(size_t)e * D_STATE + n]);
    hs[n] = 0.0f;
  }
  const float De = Dp[e];
  for (int t = 0; t < SEQLEN; ++t) {
    const int m = b * SEQLEN + t;
    float d = dtraw[(size_t)m * D_INNER + e];
    d = (d > 20.0f) ? d : log1pf(expf(d));           // softplus
    const float xv = xbs[(size_t)m * D_INNER + e];
    const float* bc = dbc + (size_t)m * (DT_RANK + 2 * D_STATE);
    const float dxv = d * xv;
    float y = 0.0f;
#pragma unroll
    for (int n = 0; n < D_STATE; ++n) {
      const float dA = expf(d * Ae[n]);
      hs[n] = dA * hs[n] + dxv * bc[DT_RANK + n];    // B slice
      y += hs[n] * bc[DT_RANK + D_STATE + n];        // C slice
    }
    y += De * xv;
    const float zv = xz[(size_t)m * (2 * D_INNER) + D_INNER + e];
    y *= zv / (1.0f + expf(-zv));                    // gate: silu(z)
    y16[(size_t)m * D_INNER + e] = (_Float16)y;
  }
}

// last timestep of each batch -> f16 (16 rows, zero padded)
__global__ void gather_last_kernel(const float* __restrict__ h,
                                   _Float16* __restrict__ hl) {
  int idx = blockIdx.x * blockDim.x + threadIdx.x;   // 16*512
  if (idx >= 16 * D_MODEL) return;
  const int r = idx >> 9, c = idx & (D_MODEL - 1);
  hl[idx] = (r < BATCH)
              ? (_Float16)h[(size_t)(r * SEQLEN + SEQLEN - 1) * D_MODEL + c]
              : (_Float16)0.0f;
}

// =====================================================================
// Host launcher
// =====================================================================
static inline int gemm_blocks(int M, int N) {
  const int waves = ((M + 15) / 16) * (N / 32);      // 16x32 tile per wave
  return (waves + 3) / 4;                            // 4 waves / 128-thread block
}

extern "C" void kernel_launch(void* const* d_in, const int* in_sizes, int n_in,
                              void* d_out, int out_size, void* d_ws, size_t ws_size,
                              hipStream_t stream) {
  (void)in_sizes; (void)n_in; (void)out_size; (void)ws_size;

  const float* x         = (const float*)d_in[0];
  const float* embed_w   = (const float*)d_in[1];
  const float* embed_b   = (const float*)d_in[2];
  const float* norm_w    = (const float*)d_in[3];
  const float* in_proj_w = (const float*)d_in[4];
  const float* conv_w    = (const float*)d_in[5];
  const float* conv_b    = (const float*)d_in[6];
  const float* x_proj_w  = (const float*)d_in[7];
  const float* dt_proj_w = (const float*)d_in[8];
  const float* dt_proj_b = (const float*)d_in[9];
  const float* A_log     = (const float*)d_in[10];
  const float* Dp        = (const float*)d_in[11];
  const float* out_proj_w= (const float*)d_in[12];
  const float* head_w    = (const float*)d_in[13];
  const float* head_b    = (const float*)d_in[14];
  float* out = (float*)d_out;

  // ---- carve workspace (256B aligned) ----
  char* base = (char*)d_ws;
  size_t off = 0;
  auto take = [&](size_t bytes) -> char* {
    char* p = base + off;
    off = (off + bytes + 255) & ~(size_t)255;
    return p;
  };
  _Float16* u16      = (_Float16*)take((size_t)MROWS * IN_DIM * 2);
  float*    h_buf    = (float*)   take((size_t)MROWS * D_MODEL * 4);
  _Float16* hn16     = (_Float16*)take((size_t)MROWS * D_MODEL * 2);
  float*    xz       = (float*)   take((size_t)MROWS * 2 * D_INNER * 4);
  float*    xbs      = (float*)   take((size_t)MROWS * D_INNER * 4);
  _Float16* xb16     = (_Float16*)take((size_t)MROWS * D_INNER * 2);
  float*    dbc      = (float*)   take((size_t)MROWS * (DT_RANK + 2 * D_STATE) * 4);
  _Float16* dbc16    = (_Float16*)take((size_t)MROWS * DT_RANK * 2);
  float*    dtraw    = (float*)   take((size_t)MROWS * D_INNER * 4);
  _Float16* y16      = (_Float16*)take((size_t)MROWS * D_INNER * 2);
  _Float16* hlast16  = (_Float16*)take((size_t)16 * D_MODEL * 2);
  _Float16* embed16  = (_Float16*)take((size_t)D_MODEL * IN_DIM * 2);
  _Float16* inproj16 = (_Float16*)take((size_t)N_LAYERS * 2 * D_INNER * D_MODEL * 2);
  _Float16* xproj16  = (_Float16*)take((size_t)N_LAYERS * (DT_RANK + 2 * D_STATE) * D_INNER * 2);
  _Float16* dtproj16 = (_Float16*)take((size_t)N_LAYERS * D_INNER * DT_RANK * 2);
  _Float16* outproj16= (_Float16*)take((size_t)N_LAYERS * D_MODEL * D_INNER * 2);
  _Float16* head16   = (_Float16*)take((size_t)HEAD_OUT * D_MODEL * 2);

  // ---- convert weights to f16 ----
  auto cvt = [&](const float* src, _Float16* dst, int n) {
    f32_to_f16_kernel<<<(n + 255) / 256, 256, 0, stream>>>(src, dst, n);
  };
  cvt(embed_w,    embed16,   D_MODEL * IN_DIM);
  cvt(in_proj_w,  inproj16,  N_LAYERS * 2 * D_INNER * D_MODEL);
  cvt(x_proj_w,   xproj16,   N_LAYERS * (DT_RANK + 2 * D_STATE) * D_INNER);
  cvt(dt_proj_w,  dtproj16,  N_LAYERS * D_INNER * DT_RANK);
  cvt(out_proj_w, outproj16, N_LAYERS * D_MODEL * D_INNER);
  cvt(head_w,     head16,    HEAD_OUT * D_MODEL);

  // ---- embed: h = u @ embed_w^T + embed_b ----
  {
    const int n = BATCH * SEQLEN * IN_DIM;
    prepare_u_kernel<<<(n + 255) / 256, 256, 0, stream>>>(x, u16);
  }
  wmma_gemm_f16<<<gemm_blocks(MROWS, D_MODEL), 128, 0, stream>>>(
      u16, embed16, h_buf, embed_b, nullptr, MROWS, D_MODEL, IN_DIM);

  // ---- layers ----
  for (int i = 0; i < N_LAYERS; ++i) {
    const _Float16* inw = inproj16  + (size_t)i * 2 * D_INNER * D_MODEL;
    const _Float16* xw  = xproj16   + (size_t)i * (DT_RANK + 2 * D_STATE) * D_INNER;
    const _Float16* dtw = dtproj16  + (size_t)i * D_INNER * DT_RANK;
    const _Float16* ow  = outproj16 + (size_t)i * D_MODEL * D_INNER;
    const float* cwp  = conv_w    + (size_t)i * D_INNER * D_CONV;
    const float* cbp  = conv_b    + (size_t)i * D_INNER;
    const float* dtbp = dt_proj_b + (size_t)i * D_INNER;
    const float* alp  = A_log     + (size_t)i * D_INNER * D_STATE;
    const float* dpp  = Dp        + (size_t)i * D_INNER;
    const float* nwp  = norm_w    + (size_t)i * D_MODEL;

    rmsnorm_kernel<<<MROWS, 256, 0, stream>>>(h_buf, nwp, hn16);

    wmma_gemm_f16<<<gemm_blocks(MROWS, 2 * D_INNER), 128, 0, stream>>>(
        hn16, inw, xz, nullptr, nullptr, MROWS, 2 * D_INNER, D_MODEL);

    {
      const int n = MROWS * D_INNER;
      conv_silu_kernel<<<(n + 255) / 256, 256, 0, stream>>>(xz, cwp, cbp, xbs, xb16);
    }

    wmma_gemm_f16<<<gemm_blocks(MROWS, DT_RANK + 2 * D_STATE), 128, 0, stream>>>(
        xb16, xw, dbc, nullptr, nullptr, MROWS, DT_RANK + 2 * D_STATE, D_INNER);

    {
      const int n = MROWS * DT_RANK;
      slice_dt_kernel<<<(n + 255) / 256, 256, 0, stream>>>(dbc, dbc16);
    }

    wmma_gemm_f16<<<gemm_blocks(MROWS, D_INNER), 128, 0, stream>>>(
        dbc16, dtw, dtraw, dtbp, nullptr, MROWS, D_INNER, DT_RANK);

    scan_kernel<<<(BATCH * D_INNER + 255) / 256, 256, 0, stream>>>(
        dtraw, xbs, dbc, xz, alp, dpp, y16);

    // h += y @ out_proj^T   (residual fused into epilogue)
    wmma_gemm_f16<<<gemm_blocks(MROWS, D_MODEL), 128, 0, stream>>>(
        y16, ow, h_buf, nullptr, h_buf, MROWS, D_MODEL, D_INNER);
  }

  // ---- head: out = h_last @ head_w^T + head_b ----
  {
    const int n = 16 * D_MODEL;
    gather_last_kernel<<<(n + 255) / 256, 256, 0, stream>>>(h_buf, hlast16);
  }
  wmma_gemm_f16<<<gemm_blocks(BATCH, HEAD_OUT), 128, 0, stream>>>(
      hlast16, head16, out, head_b, nullptr, BATCH, HEAD_OUT, D_MODEL);
}
